// CrossAttentionBlock_48498770706509
// MI455X (gfx1250) — matrix operand order repaired and checked by reading, compiled, not run on hardware
//
#include <hip/hip_runtime.h>
#include <hip/hip_bf16.h>
#include <hip/hip_fp16.h>

typedef __attribute__((ext_vector_type(16))) _Float16 v16h;
typedef __attribute__((ext_vector_type(8)))  _Float16 v8h;
typedef __attribute__((ext_vector_type(8)))  float    v8f;

#define D_MODEL 1024
#define NH      16
#define HDIM    64
#define FF_DIM  4096
#define BATCH   8
#define NQ      512
#define NKV     1024

// ---------------------------------------------------------------------------
// WMMA helpers (CDNA5 16x16x32 f16 -> f32)
// ---------------------------------------------------------------------------
__device__ __forceinline__ v8f wmma_f16(v16h a, v16h b, v8f c) {
  return __builtin_amdgcn_wmma_f32_16x16x32_f16(
      /*neg_a=*/false, a, /*neg_b=*/false, b,
      /*c_mod=*/(short)0, c, /*reuse_a=*/false, /*reuse_b=*/false);
}

// A-matrix 16x32 f16 operand (ISA 7.12.2): lane<16 holds K 0-7 & 16-23,
// lane>=16 holds K 8-15 & 24-31, row M = lane&15. p points at tile (row0,k0),
// ld = row stride in halfs. Works for global or LDS pointers.
__device__ __forceinline__ v16h load_matA(const _Float16* p, size_t ld) {
  const int lane = threadIdx.x & 31;
  const _Float16* q = p + (size_t)(lane & 15) * ld + ((lane >> 4) << 3);
  v8h lo = *(const v8h*)q;          // K k..k+7      -> VGPR 0-3
  v8h hi = *(const v8h*)(q + 16);   // K k+16..k+23  -> VGPR 4-7
  v16h r;
#pragma unroll
  for (int i = 0; i < 8; ++i) { r[i] = lo[i]; r[i + 8] = hi[i]; }
  return r;
}

// B-matrix 32x16 f16 operand: lane n (0-15) holds column n, K 0-7 & 16-23;
// lane n+16 holds K 8-15 & 24-31. With B given as W[N][K] row-major (x @ W^T), lane
// reads row n of W, K-contiguous.
__device__ __forceinline__ v16h load_matB(const _Float16* p, size_t ld) {
  const int lane = threadIdx.x & 31;
  const _Float16* q = p + (size_t)(lane & 15) * ld + ((lane >> 4) << 3);
  v8h lo = *(const v8h*)q;
  v8h hi = *(const v8h*)(q + 16);
  v16h r;
#pragma unroll
  for (int i = 0; i < 8; ++i) { r[i] = lo[i]; r[i + 8] = hi[i]; }
  return r;
}

// ---------------------------------------------------------------------------
// Elementwise utility kernels
// ---------------------------------------------------------------------------
__global__ void zero_f32_kernel(float* __restrict__ p, int n) {
  int i = blockIdx.x * blockDim.x + threadIdx.x;
  if (i < n) p[i] = 0.0f;
}

__global__ void f32_to_f16_kernel(const float* __restrict__ src,
                                  _Float16* __restrict__ dst, int n) {
  int i = blockIdx.x * blockDim.x + threadIdx.x;
  if (i < n) dst[i] = (_Float16)src[i];
}

// ---------------------------------------------------------------------------
// LayerNorm over rows of length 1024; 256 threads per row; f16 output.
// ---------------------------------------------------------------------------
__global__ __launch_bounds__(256) void layernorm_f16_kernel(
    const float* __restrict__ x, const float* __restrict__ gamma,
    const float* __restrict__ beta, _Float16* __restrict__ out) {
  const int row = blockIdx.x;
  const float* xr = x + (size_t)row * D_MODEL;
  float v[4], s = 0.0f, s2 = 0.0f;
#pragma unroll
  for (int i = 0; i < 4; ++i) {
    v[i] = xr[threadIdx.x + i * 256];
    s += v[i]; s2 += v[i] * v[i];
  }
#pragma unroll
  for (int off = 16; off; off >>= 1) {
    s  += __shfl_xor(s,  off, 32);
    s2 += __shfl_xor(s2, off, 32);
  }
  __shared__ float ws1[8], ws2[8];
  const int wave = threadIdx.x >> 5, lane = threadIdx.x & 31;
  if (lane == 0) { ws1[wave] = s; ws2[wave] = s2; }
  __syncthreads();
  float ts = 0.0f, ts2 = 0.0f;
#pragma unroll
  for (int i = 0; i < 8; ++i) { ts += ws1[i]; ts2 += ws2[i]; }
  const float mu   = ts  * (1.0f / D_MODEL);
  const float var  = ts2 * (1.0f / D_MODEL) - mu * mu;
  const float rstd = rsqrtf(var + 1e-5f);
#pragma unroll
  for (int i = 0; i < 4; ++i) {
    int c = threadIdx.x + i * 256;
    out[(size_t)row * D_MODEL + c] =
        (_Float16)((v[i] - mu) * rstd * gamma[c] + beta[c]);
  }
}

// ---------------------------------------------------------------------------
// Generic f16 WMMA GEMM:  C[M,N] = act(A[M,K] @ W[N,K]^T + bias) * scale + res
// Block = 256 thr (8 waves), block tile 64(M) x 128(N), wave tile 32x32.
// Outputs (nullable): outF fp32, outH f16 (same layout), outHT f16 transposed
// per-head V layout: [b][n][kv] with kv = m%1024.
// ---------------------------------------------------------------------------
__global__ __launch_bounds__(256) void gemm_f16_kernel(
    const _Float16* __restrict__ A, const _Float16* __restrict__ W,
    const float* __restrict__ bias, const float* __restrict__ res,
    float* __restrict__ outF, _Float16* __restrict__ outH,
    _Float16* __restrict__ outHT, int M, int N, int K, float scale, int gelu) {
  const int lane = threadIdx.x & 31;
  const int wave = threadIdx.x >> 5;
  const int m0 = blockIdx.y * 64 + (wave & 1) * 32;
  const int n0 = blockIdx.x * 128 + (wave >> 1) * 32;

  v8f acc00 = {}, acc01 = {}, acc10 = {}, acc11 = {};
  const _Float16* Ar0 = A + (size_t)m0 * K;
  const _Float16* Ar1 = A + (size_t)(m0 + 16) * K;
  const _Float16* Wr0 = W + (size_t)n0 * K;
  const _Float16* Wr1 = W + (size_t)(n0 + 16) * K;

  for (int k0 = 0; k0 < K; k0 += 32) {
    v16h a0 = load_matA(Ar0 + k0, K);
    v16h a1 = load_matA(Ar1 + k0, K);
    v16h b0 = load_matB(Wr0 + k0, K);
    v16h b1 = load_matB(Wr1 + k0, K);
    acc00 = wmma_f16(a0, b0, acc00);
    acc01 = wmma_f16(a0, b1, acc01);
    acc10 = wmma_f16(a1, b0, acc10);
    acc11 = wmma_f16(a1, b1, acc11);
  }

  const int col = lane & 15;
  const int rb  = (lane >> 4) * 8;
  v8f accs[2][2] = {{acc00, acc01}, {acc10, acc11}};
#pragma unroll
  for (int mi = 0; mi < 2; ++mi) {
#pragma unroll
    for (int ni = 0; ni < 2; ++ni) {
      const int n = n0 + ni * 16 + col;
      const float bv = bias[n];
#pragma unroll
      for (int r = 0; r < 8; ++r) {
        const int m = m0 + mi * 16 + rb + r;
        float v = accs[mi][ni][r] + bv;
        if (gelu) v = 0.5f * v * (1.0f + erff(v * 0.70710678118f));
        v *= scale;
        if (res)  v += res[(size_t)m * N + n];
        if (outF) outF[(size_t)m * N + n] = v;
        if (outH) outH[(size_t)m * N + n] = (_Float16)v;
        if (outHT) {
          const int bb = m >> 10, kv = m & 1023;   // M = B*NKV rows
          outHT[((size_t)(bb << 10) + n) * NKV + kv] = (_Float16)v;
        }
      }
    }
  }
}

// ---------------------------------------------------------------------------
// Attention: one block per (b, h, 16 q-rows). 8 waves, dynamic LDS.
//   S  : 16 x 1040 fp32 scores (padded rows vs LDS bank conflicts)
//   P  : 16 x 1024 f16 probabilities
// Scale 1/sqrt(64) is pre-folded into Qh.
// ---------------------------------------------------------------------------
#define S_LD 1040
__global__ __launch_bounds__(256) void attention_kernel(
    const _Float16* __restrict__ Qh, const _Float16* __restrict__ Kh,
    const _Float16* __restrict__ Vt, const int* __restrict__ mask,
    _Float16* __restrict__ ctxH, float* __restrict__ attnW) {
  extern __shared__ char smem[];
  float*    S = (float*)smem;                          // 16*1040*4 = 66560 B
  _Float16* P = (_Float16*)(smem + 16 * S_LD * 4);     // 16*1024*2 = 32768 B

  const int b  = blockIdx.z;
  const int h  = blockIdx.y;
  const int q0 = blockIdx.x * 16;
  const int wave = threadIdx.x >> 5, lane = threadIdx.x & 31;
  const int col = lane & 15, rb = (lane >> 4) * 8;

  // ---- Step 1: S = (Q*scale) @ K^T, masked. 64 column tiles over 8 waves.
  const _Float16* qb = Qh + ((size_t)(b * NQ + q0) * D_MODEL + h * HDIM);
  for (int ni = wave; ni < 64; ni += 8) {
    const _Float16* kb = Kh + ((size_t)(b * NKV + ni * 16) * D_MODEL + h * HDIM);
    v8f c = {};
    c = wmma_f16(load_matA(qb, D_MODEL),       load_matB(kb, D_MODEL),       c);
    c = wmma_f16(load_matA(qb + 32, D_MODEL),  load_matB(kb + 32, D_MODEL),  c);
    const int kv = ni * 16 + col;
    const bool masked = mask[(size_t)b * NKV + kv] != 0;
#pragma unroll
    for (int r = 0; r < 8; ++r)
      S[(rb + r) * S_LD + kv] = masked ? -3.0e38f : c[r];
  }
  __syncthreads();

  // ---- Step 2: row softmax; p -> P (f16) and mean-over-heads accumulation.
#pragma unroll
  for (int rr = 0; rr < 2; ++rr) {
    const int row = wave * 2 + rr;
    float* srow = S + row * S_LD;
    float mx = -3.4e38f;
    for (int k = lane; k < NKV; k += 32) mx = fmaxf(mx, srow[k]);
#pragma unroll
    for (int off = 16; off; off >>= 1) mx = fmaxf(mx, __shfl_xor(mx, off, 32));
    float sum = 0.0f;
    for (int k = lane; k < NKV; k += 32) {
      float e = __expf(srow[k] - mx);
      srow[k] = e;
      sum += e;
    }
#pragma unroll
    for (int off = 16; off; off >>= 1) sum += __shfl_xor(sum, off, 32);
    const float inv = 1.0f / sum;
    float* aw = attnW + ((size_t)(b * NQ + q0 + row) * NKV);
    for (int k = lane; k < NKV; k += 32) {
      const float p = srow[k] * inv;
      P[row * NKV + k] = (_Float16)p;
      atomicAdd(&aw[k], p * (1.0f / NH));   // mean over 16 heads
    }
  }
  __syncthreads();

  // ---- Step 3: ctx = P @ V (V stored transposed: [b][d][kv], K-contiguous).
  if (wave < 4) {
    const int ni = wave;                     // 4 x 16 output columns (HDIM=64)
    v8f c = {};
    const _Float16* vb =
        Vt + ((size_t)(b * D_MODEL + h * HDIM + ni * 16)) * NKV;
    for (int k0 = 0; k0 < NKV; k0 += 32)
      c = wmma_f16(load_matA(P + k0, NKV), load_matB(vb + k0, NKV), c);
#pragma unroll
    for (int r = 0; r < 8; ++r) {
      const int q = q0 + rb + r;
      const int d = h * HDIM + ni * 16 + col;
      ctxH[((size_t)(b * NQ + q)) * D_MODEL + d] = (_Float16)c[r];
    }
  }
}

// ---------------------------------------------------------------------------
// Launch
// ---------------------------------------------------------------------------
extern "C" void kernel_launch(void* const* d_in, const int* in_sizes, int n_in,
                              void* d_out, int out_size, void* d_ws, size_t ws_size,
                              hipStream_t stream) {
  const float* query      = (const float*)d_in[0];
  const float* key_value  = (const float*)d_in[1];
  const int*   kp_mask    = (const int*)d_in[2];
  const float* nq_gamma   = (const float*)d_in[3];
  const float* nq_beta    = (const float*)d_in[4];
  const float* nkv_gamma  = (const float*)d_in[5];
  const float* nkv_beta   = (const float*)d_in[6];
  const float* in_proj_w  = (const float*)d_in[7];
  const float* in_proj_b  = (const float*)d_in[8];
  const float* out_w      = (const float*)d_in[9];
  const float* out_b      = (const float*)d_in[10];
  const float* nff_gamma  = (const float*)d_in[11];
  const float* nff_beta   = (const float*)d_in[12];
  const float* ff1_w      = (const float*)d_in[13];
  const float* ff1_b      = (const float*)d_in[14];
  const float* ff2_w      = (const float*)d_in[15];
  const float* ff2_b      = (const float*)d_in[16];

  char* ws = (char*)d_ws;
  size_t off = 0;
  auto take = [&](size_t bytes) { char* p = ws + off; off += (bytes + 255) & ~(size_t)255; return p; };

  const size_t MQ = (size_t)BATCH * NQ;     // 4096 query rows
  const size_t MK = (size_t)BATCH * NKV;    // 8192 kv rows

  _Float16* q_in_h  = (_Float16*)take(MQ * D_MODEL * 2);
  _Float16* kv_in_h = (_Float16*)take(MK * D_MODEL * 2);
  _Float16* wproj_h = (_Float16*)take((size_t)3 * D_MODEL * D_MODEL * 2);
  _Float16* outw_h  = (_Float16*)take((size_t)D_MODEL * D_MODEL * 2);
  _Float16* ff1_h   = (_Float16*)take((size_t)FF_DIM * D_MODEL * 2);
  _Float16* ff2_h   = (_Float16*)take((size_t)D_MODEL * FF_DIM * 2);
  _Float16* Qh      = (_Float16*)take(MQ * D_MODEL * 2);
  _Float16* Kh      = (_Float16*)take(MK * D_MODEL * 2);
  _Float16* Vt      = (_Float16*)take(MK * D_MODEL * 2);
  _Float16* ctx_h   = (_Float16*)take(MQ * D_MODEL * 2);
  float*    x_ws    = (float*)   take(MQ * D_MODEL * 4);
  _Float16* hln_h   = (_Float16*)take(MQ * D_MODEL * 2);
  _Float16* hff_h   = (_Float16*)take(MQ * FF_DIM * 2);
  (void)ws_size; (void)n_in; (void)in_sizes; (void)out_size;

  float* xout  = (float*)d_out;                     // (B,Nq,D) fp32
  float* attnW = xout + MQ * D_MODEL;               // (B,Nq,Nkv) fp32

  // 0) zero attn-weights accumulator
  {
    int n = (int)(MQ * NKV);
    zero_f32_kernel<<<(n + 255) / 256, 256, 0, stream>>>(attnW, n);
  }
  // 1) weights -> f16
  {
    int n;
    n = 3 * D_MODEL * D_MODEL;
    f32_to_f16_kernel<<<(n + 255) / 256, 256, 0, stream>>>(in_proj_w, wproj_h, n);
    n = D_MODEL * D_MODEL;
    f32_to_f16_kernel<<<(n + 255) / 256, 256, 0, stream>>>(out_w, outw_h, n);
    n = FF_DIM * D_MODEL;
    f32_to_f16_kernel<<<(n + 255) / 256, 256, 0, stream>>>(ff1_w, ff1_h, n);
    n = D_MODEL * FF_DIM;
    f32_to_f16_kernel<<<(n + 255) / 256, 256, 0, stream>>>(ff2_w, ff2_h, n);
  }
  // 2) input LayerNorms
  layernorm_f16_kernel<<<(int)MQ, 256, 0, stream>>>(query, nq_gamma, nq_beta, q_in_h);
  layernorm_f16_kernel<<<(int)MK, 256, 0, stream>>>(key_value, nkv_gamma, nkv_beta, kv_in_h);

  // 3) QKV projections (scale 1/sqrt(64) folded into Q; V stored transposed)
  gemm_f16_kernel<<<dim3(D_MODEL / 128, MQ / 64), 256, 0, stream>>>(
      q_in_h, wproj_h, in_proj_b, nullptr, nullptr, Qh, nullptr,
      (int)MQ, D_MODEL, D_MODEL, 0.125f, 0);
  gemm_f16_kernel<<<dim3(D_MODEL / 128, MK / 64), 256, 0, stream>>>(
      kv_in_h, wproj_h + (size_t)D_MODEL * D_MODEL, in_proj_b + D_MODEL,
      nullptr, nullptr, Kh, nullptr, (int)MK, D_MODEL, D_MODEL, 1.0f, 0);
  gemm_f16_kernel<<<dim3(D_MODEL / 128, MK / 64), 256, 0, stream>>>(
      kv_in_h, wproj_h + (size_t)2 * D_MODEL * D_MODEL, in_proj_b + 2 * D_MODEL,
      nullptr, nullptr, nullptr, Vt, (int)MK, D_MODEL, D_MODEL, 1.0f, 0);

  // 4) attention (softmax + mean-over-heads + ctx), 99328 B dynamic LDS
  {
    const size_t smem = (size_t)16 * S_LD * 4 + (size_t)16 * NKV * 2;
    attention_kernel<<<dim3(NQ / 16, NH, BATCH), 256, smem, stream>>>(
        Qh, Kh, Vt, kp_mask, ctx_h, attnW);
  }

  // 5) out projection + residual -> x (fp32 in ws)
  gemm_f16_kernel<<<dim3(D_MODEL / 128, MQ / 64), 256, 0, stream>>>(
      ctx_h, outw_h, out_b, query, x_ws, nullptr, nullptr,
      (int)MQ, D_MODEL, D_MODEL, 1.0f, 0);

  // 6) FFN: LN -> GEMM+GELU -> GEMM + residual -> final x
  layernorm_f16_kernel<<<(int)MQ, 256, 0, stream>>>(x_ws, nff_gamma, nff_beta, hln_h);
  gemm_f16_kernel<<<dim3(FF_DIM / 128, MQ / 64), 256, 0, stream>>>(
      hln_h, ff1_h, ff1_b, nullptr, nullptr, hff_h, nullptr,
      (int)MQ, FF_DIM, D_MODEL, 1.0f, 1);
  gemm_f16_kernel<<<dim3(D_MODEL / 128, MQ / 64), 256, 0, stream>>>(
      hff_h, ff2_h, ff2_b, x_ws, xout, nullptr, nullptr,
      (int)MQ, D_MODEL, FF_DIM, 1.0f, 0);
}
